// RNN_5660766896611
// MI455X (gfx1250) — compile-verified, matrix operand order checked by reading, not run on hardware
//
#include <hip/hip_runtime.h>

// ---------------------------------------------------------------------------
// 2-layer ReLU RNN (T=1024, B=256, NIN=64, H=256) fused into ONE kernel.
// Batch rows are independent through the recurrence -> 16 blocks, each owning
// a 16-row batch tile, run the full 1024-step scan locally on one WGP using
// v_wmma_f32_16x16x32_bf16. W_ih0/W_hh0/W_hh1 fragments live in VGPRs for the
// whole scan; W_ih1 is staged to LDS (bf16, bank-padded) and streamed; the x
// tile is double-buffered through LDS with packed bf16 conversion.
// ---------------------------------------------------------------------------

typedef __attribute__((ext_vector_type(16))) __bf16       v16bf;
typedef __attribute__((ext_vector_type(8)))  float        v8f;
typedef __attribute__((ext_vector_type(4)))  unsigned int v4u;

#define T_STEPS 1024
#define BATCH   256
#define NIN     64
#define HDIM    256
#define LDH     264   // padded bf16 row stride: 132 dwords -> +4 banks per row
#define LDX     72    // padded x row stride:     36 dwords -> +36 banks per row
#define BTILE   16
#define WAVES   16
#define BLOCK_THREADS (WAVES * 32)

__device__ __forceinline__ unsigned short f2bf(float f) {
#if defined(__has_builtin) && __has_builtin(__builtin_amdgcn_cvt_pk_bf16_f32)
  typedef __attribute__((ext_vector_type(2))) __bf16 v2bf;
  v2bf p = __builtin_amdgcn_cvt_pk_bf16_f32(f, f);
  return (unsigned short)(__builtin_bit_cast(unsigned, p) & 0xFFFFu);
#else
  union { float f; unsigned u; } v; v.f = f;
  unsigned r = v.u + 0x7FFFu + ((v.u >> 16) & 1u);   // round-to-nearest-even
  return (unsigned short)(r >> 16);
#endif
}

__device__ __forceinline__ unsigned pack_bf16x2(float lo, float hi) {
#if defined(__has_builtin) && __has_builtin(__builtin_amdgcn_cvt_pk_bf16_f32)
  typedef __attribute__((ext_vector_type(2))) __bf16 v2bf;
  return __builtin_bit_cast(unsigned, __builtin_amdgcn_cvt_pk_bf16_f32(lo, hi));
#else
  return (unsigned)f2bf(lo) | ((unsigned)f2bf(hi) << 16);
#endif
}

__device__ __forceinline__ float bf2f(unsigned short s) {
  union { unsigned u; float f; } v; v.u = ((unsigned)s) << 16; return v.f;
}

union FragU { v16bf v; unsigned short s[16]; v4u q[2]; };

// A/B fragment from row-major fp32 memory (one row per lane), setup only.
// ISA 16-bit layout: elems 0..7 = K {kb+khalf+0..7}, elems 8..15 = +16.
__device__ __forceinline__ v16bf frag_from_f32(const float* __restrict__ p) {
  FragU u;
#pragma unroll
  for (int j = 0; j < 8; ++j) {
    u.s[j]     = f2bf(p[j]);
    u.s[8 + j] = f2bf(p[16 + j]);
  }
  return u.v;
}

// Fragment from LDS bf16: two 16B ds loads.
__device__ __forceinline__ v16bf frag_from_lds(const unsigned short* p) {
  FragU u;
  u.q[0] = *(const v4u*)(p);
  u.q[1] = *(const v4u*)(p + 16);
  return u.v;
}

__device__ __forceinline__ v8f wmma_bf16(v16bf a, v16bf b, v8f c) {
  return __builtin_amdgcn_wmma_f32_16x16x32_bf16(
      /*neg_a=*/false, a, /*neg_b=*/false, b,
      /*c_mod=*/(short)0, c, /*reuse_a=*/false, /*reuse_b=*/false);
}

__global__ __launch_bounds__(BLOCK_THREADS, 1)
void rnn2_fused(const float* __restrict__ x,
                const float* __restrict__ W_ih0, const float* __restrict__ b_ih0,
                const float* __restrict__ W_hh0, const float* __restrict__ b_hh0,
                const float* __restrict__ W_ih1, const float* __restrict__ b_ih1,
                const float* __restrict__ W_hh1, const float* __restrict__ b_hh1,
                const float* __restrict__ Wp,    const float* __restrict__ bp,
                float* __restrict__ out) {
  extern __shared__ unsigned short smem[];
  unsigned short* ldsWih1 = smem;                      // [HDIM][LDH]    bf16
  unsigned short* ldsH0   = smem + HDIM * LDH;         // [BTILE][LDH]   bf16
  unsigned short* ldsH1   = ldsH0 + BTILE * LDH;       // [BTILE][LDH]   bf16
  unsigned short* ldsX    = ldsH1 + BTILE * LDH;       // [2][BTILE][LDX] bf16
  unsigned*       ldsX32  = (unsigned*)ldsX;

  const int tid   = threadIdx.x;
  const int lane  = tid & 31;
  const int wave  = tid >> 5;            // 0..15, owns cols [16w,16w+16)
  const int b0    = blockIdx.x * BTILE;  // batch tile base
  const int n0    = wave * 16;
  const int col   = lane & 15;           // N (output column) for B/C frags
  const int row   = lane & 15;           // M (row) for A frags
  const int khalf = (lane >> 4) * 8;     // lane-half K offset

  // Stage W_ih1 -> LDS as bf16 (streamed B-frags for layer-1 input GEMM).
  for (int idx = tid; idx < (HDIM * HDIM) / 2; idx += BLOCK_THREADS) {
    int r = idx >> 7, c = (idx & 127) * 2;
    const float* wsrc = W_ih1 + r * HDIM + c;
    *(unsigned*)&ldsWih1[r * LDH + c] = pack_bf16x2(wsrc[0], wsrc[1]);
  }
  // Zero initial hidden state h0, h1.
  for (int idx = tid; idx < 2 * BTILE * LDH; idx += BLOCK_THREADS)
    ldsH0[idx] = 0;
  // Stage x tile for t=0 into buffer 0 (thread -> one float2).
  {
    int r = tid >> 5, c = (tid & 31) * 2;
    const float* xs = x + (size_t)(b0 + r) * NIN + c;
    ldsX32[(r * LDX + c) >> 1] = pack_bf16x2(xs[0], xs[1]);
  }

  // VGPR-resident weight B-fragments (held for all 1024 steps).
  v16bf fWih0[2], fWhh0[8], fWhh1[8];
  {
    const float* rp0 = W_ih0 + (size_t)(n0 + col) * NIN + khalf;
#pragma unroll
    for (int kk = 0; kk < 2; ++kk) fWih0[kk] = frag_from_f32(rp0 + kk * 32);
    const float* rp1 = W_hh0 + (size_t)(n0 + col) * HDIM + khalf;
#pragma unroll
    for (int kk = 0; kk < 8; ++kk) fWhh0[kk] = frag_from_f32(rp1 + kk * 32);
    const float* rp2 = W_hh1 + (size_t)(n0 + col) * HDIM + khalf;
#pragma unroll
    for (int kk = 0; kk < 8; ++kk) fWhh1[kk] = frag_from_f32(rp2 + kk * 32);
  }

  const float bias0 = b_ih0[n0 + col] + b_hh0[n0 + col];
  const float bias1 = b_ih1[n0 + col] + b_hh1[n0 + col];

  // Head weights: lane holds Wp[8*lane + j].
  float wp[8];
#pragma unroll
  for (int j = 0; j < 8; ++j) wp[j] = Wp[lane * 8 + j];
  const float bpv = bp[0];

  __syncthreads();

  const unsigned short* aH0 = ldsH0 + row * LDH + khalf;
  const unsigned short* aH1 = ldsH1 + row * LDH + khalf;
  const unsigned short* aX  = ldsX  + row * LDX + khalf;
  const unsigned short* bW1 = ldsWih1 + (n0 + col) * LDH + khalf;
  unsigned short* stH0 = ldsH0 + (8 * (lane >> 4)) * LDH + n0 + col;
  unsigned short* stH1 = ldsH1 + (8 * (lane >> 4)) * LDH + n0 + col;

  // Per-thread staging coordinates for the x tile.
  const int sr = tid >> 5;
  const int sc = (tid & 31) * 2;
  const float* xstage = x + (size_t)(b0 + sr) * NIN + sc;  // advance by B*NIN per step

  for (int t = 0; t < T_STEPS; ++t) {
    // ---- two independent WMMA chains, interleaved for ILP ----
    // acc0 = bias0 + x_t@W_ih0^T + h0_prev@W_hh0^T
    // acc1 = bias1 + h1_prev@W_hh1^T
    v8f acc0, acc1;
#pragma unroll
    for (int i = 0; i < 8; ++i) { acc0[i] = bias0; acc1[i] = bias1; }

    const unsigned short* xb = aX + (t & 1) * (BTILE * LDX);
#pragma unroll
    for (int kk = 0; kk < 2; ++kk)
      acc0 = wmma_bf16(frag_from_lds(xb + kk * 32), fWih0[kk], acc0);
#pragma unroll
    for (int kk = 0; kk < 8; ++kk) {
      acc0 = wmma_bf16(frag_from_lds(aH0 + kk * 32), fWhh0[kk], acc0);
      acc1 = wmma_bf16(frag_from_lds(aH1 + kk * 32), fWhh1[kk], acc1);
    }

    // Stage next x tile into the other buffer; prefetch the one after.
    if (t + 1 < T_STEPS) {
      const float* xs = xstage + (size_t)(t + 1) * BATCH * NIN;
      ldsX32[((t + 1) & 1) * (BTILE * LDX / 2) + ((sr * LDX + sc) >> 1)] =
          pack_bf16x2(xs[0], xs[1]);
      if (t + 2 < T_STEPS)
        __builtin_prefetch(xs + (size_t)BATCH * NIN, 0, 1);
    }

    unsigned short r0[8];
#pragma unroll
    for (int i = 0; i < 8; ++i) r0[i] = f2bf(fmaxf(acc0[i], 0.f));

    __syncthreads();  // all waves done reading h0_prev / h1_prev
#pragma unroll
    for (int r = 0; r < 8; ++r) stH0[r * LDH] = r0[r];   // write h0_t
    __syncthreads();  // h0_t visible to all waves

    // ---- layer 1 finish: + h0_t@W_ih1^T (B-frags streamed from LDS) ----
#pragma unroll
    for (int kk = 0; kk < 8; ++kk) {
      v16bf a = frag_from_lds(aH0 + kk * 32);
      v16bf b = frag_from_lds(bW1 + kk * 32);
      acc1 = wmma_bf16(a, b, acc1);
    }
#pragma unroll
    for (int r = 0; r < 8; ++r) stH1[r * LDH] = f2bf(fmaxf(acc1[r], 0.f));
    __syncthreads();  // h1_t visible

    // ---- sigmoid head: wave w reduces batch row w over H=256 ----
    {
      const unsigned short* hp = ldsH1 + wave * LDH + lane * 8;
      v4u q = *(const v4u*)hp;
      const unsigned short* qs = (const unsigned short*)&q;
      float s = 0.f;
#pragma unroll
      for (int j = 0; j < 8; ++j) s += wp[j] * bf2f(qs[j]);
#pragma unroll
      for (int off = 16; off > 0; off >>= 1) s += __shfl_xor(s, off, 32);
      if (lane == 0)
        out[(size_t)t * BATCH + b0 + wave] = 1.f / (1.f + __expf(-(s + bpv)));
    }
    // Next iteration's first __syncthreads orders these reads vs. overwrites.
  }
}

extern "C" void kernel_launch(void* const* d_in, const int* in_sizes, int n_in,
                              void* d_out, int out_size, void* d_ws, size_t ws_size,
                              hipStream_t stream) {
  (void)in_sizes; (void)n_in; (void)d_ws; (void)ws_size; (void)out_size;
  const float* x     = (const float*)d_in[0];
  const float* W_ih0 = (const float*)d_in[1];
  const float* b_ih0 = (const float*)d_in[2];
  const float* W_hh0 = (const float*)d_in[3];
  const float* b_hh0 = (const float*)d_in[4];
  const float* W_ih1 = (const float*)d_in[5];
  const float* b_ih1 = (const float*)d_in[6];
  const float* W_hh1 = (const float*)d_in[7];
  const float* b_hh1 = (const float*)d_in[8];
  const float* Wp    = (const float*)d_in[9];
  const float* bpp   = (const float*)d_in[10];
  float* out = (float*)d_out;

  const size_t smem_bytes =
      (size_t)(HDIM * LDH + 2 * BTILE * LDH + 2 * BTILE * LDX) *
      sizeof(unsigned short);  // ~153 KB < 320 KB/WGP (CDNA5 only)

  hipLaunchKernelGGL(rnn2_fused, dim3(BATCH / BTILE), dim3(BLOCK_THREADS),
                     smem_bytes, stream,
                     x, W_ih0, b_ih0, W_hh0, b_hh0,
                     W_ih1, b_ih1, W_hh1, b_hh1, Wp, bpp, out);
}
